// TPDUEncoder_38345468019235
// MI455X (gfx1250) — matrix operand order repaired and checked by reading, not compiled, not run on hardware
//
#include <hip/hip_runtime.h>
#include <hip/hip_bf16.h>
#include <math.h>

// ---------------------------------------------------------------------------
// Types for CDNA5 (gfx1250) wave32 WMMA
// ---------------------------------------------------------------------------
typedef __attribute__((ext_vector_type(16))) __bf16 v16bf;
typedef __attribute__((ext_vector_type(8)))  float  v8f;

#define D_DIM 1024     // feature dim
#define KT_N  32       // number of K-chunks of 32 (K = 1024)
#define T_LEN 256      // sequence length
#define B_SZ  32       // batch

// ---------------------------------------------------------------------------
// Workspace layout (byte offsets into d_ws)
// ---------------------------------------------------------------------------
#define MB (1024ull * 1024ull)
#define OFF_U      (0ull)                    // f32 1024x1024 : U  = V^T @ Wu^T
#define OFF_RT     (4ull  * MB)              // f32 1024x1024 : R^T = Wr @ V
#define OFF_M      (8ull  * MB)              // f32 1024x1024 : M  = Ww^T @ U
#define OFF_UPK    (12ull * MB)              // bf16 packed B-frags of U
#define OFF_RPK    (14ull * MB)              // bf16 packed B-frags of R^T
#define OFF_WBPK   (16ull * MB)              // bf16 packed B-frags of Wb^T
#define OFF_WXPK   (18ull * MB)              // bf16 packed B-frags of Wx^T
#define OFF_MPK    (20ull * MB)              // bf16 packed B-frags of M
#define OFF_FX     (22ull * MB)              // f32 8192x1024 : relu(seq @ M)
#define OFF_GX     (54ull * MB)              // f32 8192x1024 : seq @ Wx^T + biases
#define OFF_NUMPK  (86ull * MB)              // bf16 A-frag packed num (32x1024)
#define OFF_G      (86ull * MB + 256ull*1024)  // f32 32x1024 gate
#define OFF_PART   (86ull * MB + 512ull*1024)  // f32 64x32 row-sum partials
#define OFF_BTA    (86ull * MB + 768ull*1024)  // f32 32x1024 state ping
#define OFF_BTB    (86ull * MB + 1024ull*1024) // f32 32x1024 state pong

// ---------------------------------------------------------------------------
// Stage a 32 x 1024 f32 block into LDS as bf16 in WMMA A-fragment layout.
// 16-bit A 16x32 layout: lane = (M&15) | (((k&31)>>3 & 1) << 4),
//                        elem j: j<8 -> K = 8*(lane>>4) + j ; j>=8 -> K = 16 + 8*(lane>>4) + (j-8)
// ---------------------------------------------------------------------------
__device__ __forceinline__ void stage_A32(const float* __restrict__ src,
                                          __bf16* __restrict__ sA) {
  for (int i = threadIdx.x; i < 32 * D_DIM; i += blockDim.x) {
    int m  = i >> 10;            // row 0..31
    int k  = i & (D_DIM - 1);    // 0..1023
    int kt = k >> 5;
    int kk = k & 31;
    int laneA = (m & 15) | (((kk >> 3) & 1) << 4);
    int j     = (kk & 7) | (((kk >> 4) & 1) << 3);
    sA[((((kt << 1) | (m >> 4)) * 32 + laneA) << 4) | j] = (__bf16)src[i];
  }
}

// One 16x16 output tile: A from LDS fragments, B from packed global fragments.
__device__ __forceinline__ v8f wmma_row(const __bf16* __restrict__ sA, int mTile,
                                        const v16bf* __restrict__ Bp, int nTile) {
  v8f acc = {};
  const int lane = threadIdx.x & 31;
  const v16bf* bp = Bp + (size_t)nTile * KT_N * 32 + lane;
#pragma unroll 4
  for (int kt = 0; kt < KT_N; ++kt) {
    v16bf a = *(const v16bf*)(sA + ((((kt << 1) | mTile) * 32 + lane) << 4));
    v16bf b = bp[kt * 32];
    __builtin_prefetch(bp + (kt + 2) * 32, 0, 1);   // global_prefetch_b8
    acc = __builtin_amdgcn_wmma_f32_16x16x32_bf16(false, a, false, b,
                                                  (short)0, acc, false, false);
  }
  return acc;
}

// ---------------------------------------------------------------------------
// One-off f32 tiled GEMM (1024^3) for U, R^T, M. transA/transB select op.
// ---------------------------------------------------------------------------
__global__ void gemm_f32_tiled(const float* __restrict__ A, const float* __restrict__ B,
                               float* __restrict__ C, int transA, int transB) {
  __shared__ float As[16][17];
  __shared__ float Bs[16][17];
  const int tx = threadIdx.x & 15, ty = threadIdx.x >> 4;
  const int row = blockIdx.y * 16 + ty;
  const int col = blockIdx.x * 16 + tx;
  float acc = 0.f;
  for (int k0 = 0; k0 < D_DIM; k0 += 16) {
    As[ty][tx] = transA ? A[(size_t)(k0 + tx) * D_DIM + row]
                        : A[(size_t)row * D_DIM + (k0 + tx)];
    Bs[ty][tx] = transB ? B[(size_t)col * D_DIM + (k0 + ty)]
                        : B[(size_t)(k0 + ty) * D_DIM + col];
    __syncthreads();
#pragma unroll
    for (int kk = 0; kk < 16; ++kk) acc = fmaf(As[ty][kk], Bs[kk][tx], acc);
    __syncthreads();
  }
  C[(size_t)row * D_DIM + col] = acc;
}

// ---------------------------------------------------------------------------
// Pack a 1024x1024 f32 matrix (optionally transposed) into bf16 WMMA B-frags.
// B 32x16 layout: lane = (N&15) | (((k&31)>>3 & 1) << 4), elem j: j<8 -> K = 8*(lane>>4)+j ; j>=8 -> K = 16+8*(lane>>4)+(j-8)
// dst index p = ((nTile*32 + kt)*32 + lane)*16 + j   (16 contiguous bf16/lane)
// ---------------------------------------------------------------------------
__global__ void packB_kernel(const float* __restrict__ src, __bf16* __restrict__ dst,
                             int trans) {
  int p = blockIdx.x * blockDim.x + threadIdx.x;    // 0 .. 1M-1
  int j     = p & 15;
  int lane  = (p >> 4) & 31;
  int kt    = (p >> 9) & 31;
  int nTile = p >> 14;
  int n = (nTile << 4) | (lane & 15);
  int k = (kt << 5) | (((j >> 3) & 1) << 4) | ((lane >> 4) << 3) | (j & 7);
  float v = trans ? src[(size_t)n * D_DIM + k] : src[(size_t)k * D_DIM + n];
  dst[p] = (__bf16)v;
}

__global__ void zero_f32(float* __restrict__ p, int n) {
  int i = blockIdx.x * blockDim.x + threadIdx.x;
  if (i < n) p[i] = 0.f;
}

// ---------------------------------------------------------------------------
// Big batched GEMM over all timesteps: C = epilogue(A(8192x1024) @ Bpk)
// mode 0: relu (FX).  mode 1: + bias0[n] + bias1[n] (GX).
// Grid: (16 nBlocks, 256 mBlocks), 256 threads (8 waves, one 16x16 tile each).
// ---------------------------------------------------------------------------
__global__ void wmma_gemm_big(const float* __restrict__ A, const v16bf* __restrict__ Bp,
                              float* __restrict__ C, const float* __restrict__ bias0,
                              const float* __restrict__ bias1, int mode) {
  __shared__ __align__(32) __bf16 sA[32 * D_DIM];
  const int mRow0 = blockIdx.y * 32;
  stage_A32(A + (size_t)mRow0 * D_DIM, sA);
  __syncthreads();

  const int lane = threadIdx.x & 31, w = threadIdx.x >> 5;
  const int mTile = w >> 2;
  const int nTile = blockIdx.x * 4 + (w & 3);
  v8f acc = wmma_row(sA, mTile, Bp, nTile);

  const int n  = nTile * 16 + (lane & 15);
  const int mB = mTile * 16 + ((lane >> 4) << 3);
#pragma unroll
  for (int r = 0; r < 8; ++r) {
    int m = mRow0 + mB + r;
    float v = acc[r];
    if (mode == 0) v = fmaxf(v, 0.f);
    else           v += bias0[n] + bias1[n];
    C[(size_t)m * D_DIM + n] = v;
  }
}

// ---------------------------------------------------------------------------
// Step phase 1: P = bt @ [U | Wb^T]  (N = 2048 fused).
// U half : num = (relu(acc) + FX_t)^2 -> packed A-frags + row-sum partials.
// Wb half: g = sigmoid(acc + GX_t).
// Grid: 32 blocks x 256 threads.
// ---------------------------------------------------------------------------
__global__ void step_phase1(const float* __restrict__ bt,
                            const v16bf* __restrict__ Upk,
                            const v16bf* __restrict__ Wbpk,
                            const float* __restrict__ FXt,
                            const float* __restrict__ GXt,
                            __bf16* __restrict__ numPk,
                            float* __restrict__ gbuf,
                            float* __restrict__ partial) {
  __shared__ __align__(32) __bf16 sA[32 * D_DIM];
  stage_A32(bt, sA);
  __syncthreads();

  const int lane = threadIdx.x & 31, w = threadIdx.x >> 5;
  const int mTile = w >> 2;
  const int cTile = blockIdx.x * 4 + (w & 3);      // 0..127 over fused N=2048
  const bool isU  = (cTile < 64);
  const int nTile = isU ? cTile : (cTile - 64);
  const v16bf* Bp = isU ? Upk : Wbpk;

  v8f acc = wmma_row(sA, mTile, Bp, nTile);

  const int nl = lane & 15, hi = lane >> 4;
  const int n  = nTile * 16 + nl;
  if (isU) {
#pragma unroll
    for (int r = 0; r < 8; ++r) {
      int m = mTile * 16 + hi * 8 + r;
      float fb = fmaxf(acc[r], 0.f);
      float nm = fb + FXt[(size_t)m * D_DIM + n];
      nm = nm * nm;
      // scatter into A-fragment layout for phase 2 (k dimension = n)
      int kt = n >> 5, kk = n & 31;
      int laneA = (m & 15) | (((kk >> 3) & 1) << 4);
      int j     = (kk & 7) | (((kk >> 4) & 1) << 3);
      numPk[((((kt << 1) | (m >> 4)) * 32 + laneA) << 4) | j] = (__bf16)nm;
      // per-tile row sum across the 16 lanes sharing this row
      float t = nm;
      t += __shfl_xor(t, 1, 16);
      t += __shfl_xor(t, 2, 16);
      t += __shfl_xor(t, 4, 16);
      t += __shfl_xor(t, 8, 16);
      if (nl == 0) partial[nTile * 32 + m] = t;
    }
  } else {
#pragma unroll
    for (int r = 0; r < 8; ++r) {
      int m = mTile * 16 + hi * 8 + r;
      float x = acc[r] + GXt[(size_t)m * D_DIM + n];
      gbuf[(size_t)m * D_DIM + n] = 1.f / (1.f + __expf(-x));
    }
  }
}

// ---------------------------------------------------------------------------
// Step phase 2: ub = num @ R^T ; bt_new = g*tanh(ub/S) + (1-g)*bt_old.
// Grid: 16 blocks x 256 threads. A-frags come pre-packed from phase 1.
// ---------------------------------------------------------------------------
__global__ void step_phase2(const v16bf* __restrict__ numPk,
                            const v16bf* __restrict__ Rpk,
                            const float* __restrict__ partial,
                            const float* __restrict__ gbuf,
                            const float* __restrict__ btOld,
                            float* __restrict__ btNew) {
  __shared__ float sS[32];
  if (threadIdx.x < 32) {
    float s = 0.f;
#pragma unroll 4
    for (int nt = 0; nt < 64; ++nt) s += partial[nt * 32 + threadIdx.x];
    sS[threadIdx.x] = s;
  }
  __syncthreads();

  const int lane = threadIdx.x & 31, w = threadIdx.x >> 5;
  const int mTile = w >> 2;
  const int nTile = blockIdx.x * 4 + (w & 3);

  v8f acc = {};
  const v16bf* ap = numPk + lane;
  const v16bf* bp = Rpk + (size_t)nTile * KT_N * 32 + lane;
#pragma unroll 4
  for (int kt = 0; kt < KT_N; ++kt) {
    v16bf a = ap[((kt << 1) | mTile) * 32];
    v16bf b = bp[kt * 32];
    __builtin_prefetch(bp + (kt + 2) * 32, 0, 1);
    acc = __builtin_amdgcn_wmma_f32_16x16x32_bf16(false, a, false, b,
                                                  (short)0, acc, false, false);
  }

  const int nl = lane & 15, hi = lane >> 4;
  const int n  = nTile * 16 + nl;
#pragma unroll
  for (int r = 0; r < 8; ++r) {
    int m = mTile * 16 + hi * 8 + r;
    float b   = acc[r] / sS[m];
    float g   = gbuf[(size_t)m * D_DIM + n];
    float old = btOld[(size_t)m * D_DIM + n];
    btNew[(size_t)m * D_DIM + n] = g * tanhf(b) + (1.f - g) * old;
  }
}

// ---------------------------------------------------------------------------
// Host-side orchestration (all on `stream`; graph-capture safe)
// ---------------------------------------------------------------------------
extern "C" void kernel_launch(void* const* d_in, const int* in_sizes, int n_in,
                              void* d_out, int out_size, void* d_ws, size_t ws_size,
                              hipStream_t stream) {
  const float* seq  = (const float*)d_in[0];  // (256, 32, 1024)
  const float* Wu   = (const float*)d_in[1];
  const float* Wr   = (const float*)d_in[2];
  const float* Wb   = (const float*)d_in[3];
  const float* Wb_b = (const float*)d_in[4];
  const float* Wx   = (const float*)d_in[5];
  const float* Wx_b = (const float*)d_in[6];
  const float* Ww   = (const float*)d_in[7];
  const float* V    = (const float*)d_in[8];

  char* ws = (char*)d_ws;
  float*  U    = (float*)(ws + OFF_U);
  float*  RT   = (float*)(ws + OFF_RT);
  float*  Mw   = (float*)(ws + OFF_M);
  __bf16* Upk  = (__bf16*)(ws + OFF_UPK);
  __bf16* Rpk  = (__bf16*)(ws + OFF_RPK);
  __bf16* Wbpk = (__bf16*)(ws + OFF_WBPK);
  __bf16* Wxpk = (__bf16*)(ws + OFF_WXPK);
  __bf16* Mpk  = (__bf16*)(ws + OFF_MPK);
  float*  FX   = (float*)(ws + OFF_FX);
  float*  GX   = (float*)(ws + OFF_GX);
  __bf16* numPk = (__bf16*)(ws + OFF_NUMPK);
  float*  gbuf = (float*)(ws + OFF_G);
  float*  part = (float*)(ws + OFF_PART);
  float*  btA  = (float*)(ws + OFF_BTA);
  float*  btB  = (float*)(ws + OFF_BTB);

  const dim3 g64(64, 64), blk(256);

  // ---- one-off weight products (f32) ----
  gemm_f32_tiled<<<g64, blk, 0, stream>>>(V,  Wu, U,  1, 1);  // U  = V^T @ Wu^T
  gemm_f32_tiled<<<g64, blk, 0, stream>>>(Wr, V,  RT, 0, 0);  // R^T = Wr @ V
  gemm_f32_tiled<<<g64, blk, 0, stream>>>(Ww, U,  Mw, 1, 0);  // M  = Ww^T @ U

  // ---- pack B operands into WMMA fragment layout (bf16) ----
  packB_kernel<<<4096, blk, 0, stream>>>(U,  Upk,  0);
  packB_kernel<<<4096, blk, 0, stream>>>(RT, Rpk,  0);
  packB_kernel<<<4096, blk, 0, stream>>>(Mw, Mpk,  0);
  packB_kernel<<<4096, blk, 0, stream>>>(Wb, Wbpk, 1);  // Wb^T
  packB_kernel<<<4096, blk, 0, stream>>>(Wx, Wxpk, 1);  // Wx^T

  // ---- hoisted batched GEMMs over all 8192 rows ----
  wmma_gemm_big<<<dim3(16, 256), blk, 0, stream>>>(seq, (const v16bf*)Mpk,  FX,
                                                   nullptr, nullptr, 0);
  wmma_gemm_big<<<dim3(16, 256), blk, 0, stream>>>(seq, (const v16bf*)Wxpk, GX,
                                                   Wx_b, Wb_b, 1);

  // ---- recurrence ----
  zero_f32<<<128, blk, 0, stream>>>(btA, B_SZ * D_DIM);
  float* cur = btA;
  float* nxt = btB;
  for (int t = 0; t < T_LEN; ++t) {
    const float* FXt = FX + (size_t)t * B_SZ * D_DIM;
    const float* GXt = GX + (size_t)t * B_SZ * D_DIM;
    step_phase1<<<32, blk, 0, stream>>>(cur, (const v16bf*)Upk, (const v16bf*)Wbpk,
                                        FXt, GXt, numPk, gbuf, part);
    float* outp = (t == T_LEN - 1) ? (float*)d_out : nxt;
    step_phase2<<<16, blk, 0, stream>>>((const v16bf*)numPk, (const v16bf*)Rpk,
                                        part, gbuf, cur, outp);
    float* tmp = cur; cur = nxt; nxt = tmp;
  }
  (void)in_sizes; (void)n_in; (void)out_size; (void)ws_size;
}
